// AttentionPPO_22325240005450
// MI455X (gfx1250) — compile-verified, matrix operand order checked by reading, not run on hardware
//
#include <hip/hip_runtime.h>
#include <hip/hip_bf16.h>
#include <math.h>

typedef __attribute__((ext_vector_type(16))) _Float16 v16h;
typedef __attribute__((ext_vector_type(8)))  float    v8f;
typedef __attribute__((ext_vector_type(4)))  unsigned int u32x4;
typedef __attribute__((ext_vector_type(8)))  int      i32x8;
typedef __attribute__((ext_vector_type(4)))  int      i32x4;

#define NW       8      // waves per workgroup (256 threads, wave32)
#define NTHREADS 256
#define SPB      8      // samples per workgroup

#if defined(__HIP_DEVICE_COMPILE__) && __has_builtin(__builtin_amdgcn_tensor_load_to_lds) && __has_builtin(__builtin_amdgcn_s_wait_tensorcnt)
#define HAVE_TDM 1
#else
#define HAVE_TDM 0
#endif

// f16 weight blob layout inside d_ws (element offsets):
//   tokW (padded 64x64) @ 0
//   block b: Win @ 4096+b*49152, Wout @ +12288, W1 @ +16384, W2 @ +32768
#define WF_TOKW   0
#define WF_BLK(b) (4096 + (b) * 49152)
#define WF_TOTAL  (4096 + 3 * 49152)          // 151552 halves = 303104 bytes

// ---------------- parameter plumbing ----------------
struct BlockP {
  const float *ln1_g, *ln1_b, *Win, *bin, *Wout, *bout;
  const float *ln2_g, *ln2_b, *W1, *b1, *W2, *b2;
};
struct PpoArgs {
  const float *obs, *tokW, *tokb, *macro_pos, *micro_pos, *fusion_pos;
  BlockP mac, mic, fus;
  const float *aW1, *ab1, *aW2, *ab2, *cW1, *cb1, *cW2, *cb2;
  float* out;
  int B;
};

// ---------------- TDM: DMA one NxK f16 weight tile from global into LDS ----------------
__device__ __forceinline__ unsigned lds_off(const void* p) {
  // LDS aperture: low 32 bits of the flat address are the LDS byte address
  return (unsigned)(unsigned long long)p;
}
__device__ __forceinline__ void tdm_load_w(const _Float16* gsrc, unsigned ldsOff,
                                           unsigned K, unsigned N) {
#if HAVE_TDM
  unsigned long long ga = (unsigned long long)gsrc;
  u32x4 g0 = { 1u,                                   // count=1 (valid user descriptor)
               ldsOff,                               // lds_addr
               (unsigned)(ga & 0xFFFFFFFFu),         // global_addr[31:0]
               (unsigned)((ga >> 32) & 0x01FFFFFFu) | (2u << 30) };  // addr[56:32] | type=2
  i32x8 g1 = { (int)(1u << 16),                      // wg_mask=0, data_size=1 (2B)
               (int)((K & 0xFFFFu) << 16),           // tensor_dim0[15:0]
               (int)((K >> 16) | ((N & 0xFFFFu) << 16)),   // dim0 hi | tensor_dim1 lo
               (int)((N >> 16) | ((K & 0xFFFFu) << 16)),   // dim1 hi | tile_dim0=K
               (int)(N & 0xFFFFu),                   // tile_dim1=N, tile_dim2=0
               (int)K,                               // tensor_dim0_stride[31:0]
               0, 0 };
  i32x4 z4 = { 0, 0, 0, 0 };
  i32x8 z8 = { 0, 0, 0, 0, 0, 0, 0, 0 };
  __builtin_amdgcn_tensor_load_to_lds(g0, g1, z4, z4, z8, 0);
#else
  (void)gsrc; (void)ldsOff; (void)K; (void)N;
#endif
}
template <int N>
__device__ __forceinline__ void tdm_wait() {
#if HAVE_TDM
  __builtin_amdgcn_s_wait_tensorcnt(N);
#endif
}

// ---------------- WMMA fragment loader (16x16x32 f16, A and B identical pattern) ----------------
__device__ __forceinline__ v16h load_frag(const _Float16* base, int ld,
                                          int r0, int k0, int lane) {
  int row = r0 + (lane & 15);
  int kg  = (lane >> 4) ? 8 : 0;
  const _Float16* p = base + row * ld + k0 + kg;
  v16h a;
#pragma unroll
  for (int v = 0; v < 4; ++v) { a[2*v] = p[2*v];      a[2*v+1] = p[2*v+1]; }
#pragma unroll
  for (int v = 4; v < 8; ++v) { int k = 16 + 2*(v-4); a[2*v] = p[k]; a[2*v+1] = p[k+1]; }
  return a;
}

// ---------------- fallback cooperative staging: global f32 -> LDS f16 ----------------
__device__ __forceinline__ void stage_w(const float* W, int n, _Float16* dst) {
#pragma clang loop unroll_count(2)
  for (int i = threadIdx.x; i < n; i += NTHREADS) dst[i] = (_Float16)W[i];
}
__device__ __forceinline__ void stage_tokw(const float* W, _Float16* dst) {
#pragma clang loop unroll_count(2)
  for (int i = threadIdx.x; i < 64 * 64; i += NTHREADS) {
    int r = i >> 6, c = i & 63;
    dst[i] = (c < 54) ? (_Float16)W[r * 54 + c] : (_Float16)0.f;
  }
}

// ---------------- generic WMMA GEMM: D = A(lds f16, MxK) * W^T(lds f16, NxK) ----------------
// MODE 0: dstF[rm,n]  = acc + bias[n]
// MODE 1: resid[(rm/L)*RPS+TOKOFF+rm%L, n] += acc + bias[n]
// MODE 2: dstH[rm,n]  = f16(gelu(acc + bias[n]))
// MODE 3: dstF[rm,n]  = acc + bias[n] + pos(token)   (embedding epilogue)
template <int MODE, int RPS, int TOKOFF, int L>
__device__ void wgemm(const _Float16* A, int lda, int M,
                      const _Float16* Bw, int ldb, int N, int K,
                      const float* bias,
                      float* dstF, int ldf, _Float16* dstH, int ldh,
                      const float* posA, const float* posB) {
  int tid  = threadIdx.x;
  int lane = tid & 31, wave = tid >> 5;
  int nM = M >> 4, nN = N >> 4, nT = nM * nN;
  for (int t = wave; t < nT; t += NW) {
    int m0 = (t / nN) << 4, n0 = (t % nN) << 4;
    v8f acc = {};
    for (int k0 = 0; k0 < K; k0 += 32) {
      v16h af = load_frag(A,  lda, m0, k0, lane);
      v16h bf = load_frag(Bw, ldb, n0, k0, lane);
      acc = __builtin_amdgcn_wmma_f32_16x16x32_f16(false, af, false, bf,
                                                   (short)0, acc, false, false);
    }
    int n = n0 + (lane & 15);
    int mbase = m0 + ((lane >> 4) << 3);
    float bn = bias[n];
#pragma unroll
    for (int v = 0; v < 8; ++v) {
      int rm = mbase + v;
      float val = acc[v] + bn;
      if (MODE == 0) {
        dstF[rm * ldf + n] = val;
      } else if (MODE == 1) {
        int s = rm / L, q = rm - s * L;
        dstF[(s * RPS + TOKOFF + q) * ldf + n] += val;
      } else if (MODE == 2) {
        float g = 0.5f * val * (1.f + erff(val * 0.70710678118654752f));
        dstH[rm * ldh + n] = (_Float16)g;
      } else { // embedding: add positional by token id
        int t8 = rm & 7;
        float p = (t8 < 2) ? posA[t8 * 64 + n] : posB[(t8 - 2) * 64 + n];
        dstF[rm * ldf + n] = val + p;
      }
    }
  }
}

// ---------------- LayerNorm rows -> f16 staging ----------------
template <int RPS, int TOKOFF, int L>
__device__ void layernorm_rows(const float* resid, int M,
                               const float* g, const float* b, _Float16* out) {
  int tid = threadIdx.x;
  if (tid < M) {
    int s = tid / L, q = tid - s * L;
    const float* x = resid + (s * RPS + TOKOFF + q) * 64;
    float mu = 0.f;
#pragma clang loop unroll(disable)
    for (int i = 0; i < 64; ++i) mu += x[i];
    mu *= (1.f / 64.f);
    float var = 0.f;
#pragma clang loop unroll(disable)
    for (int i = 0; i < 64; ++i) { float d = x[i] - mu; var += d * d; }
    var *= (1.f / 64.f);
    float inv = rsqrtf(var + 1e-5f);
#pragma clang loop unroll(disable)
    for (int i = 0; i < 64; ++i)
      out[tid * 64 + i] = (_Float16)((x[i] - mu) * inv * g[i] + b[i]);
  }
}

// ---------------- tiny softmax attention (HD=16, L<=6), VALU per (sample,head) ----------------
template <int L>
__device__ void attention(const float* qkv, float* keyavg, _Float16* ctxH) {
  int tid = threadIdx.x;
  if (tid < SPB * L) keyavg[tid] = 0.f;
  __syncthreads();
  if (tid < SPB * 4) {
    int s = tid >> 2, h = tid & 3;
    const float* base = qkv + (s * L) * 192;
    float sc[L][L];
#pragma unroll
    for (int qi = 0; qi < L; ++qi)
#pragma unroll
      for (int ki = 0; ki < L; ++ki) {
        const float* qp = base + qi * 192 + h * 16;
        const float* kp = base + ki * 192 + 64 + h * 16;
        float d = 0.f;
#pragma clang loop unroll(disable)
        for (int e = 0; e < 16; ++e) d += qp[e] * kp[e];
        sc[qi][ki] = d * 0.25f;                     // 1/sqrt(16)
      }
#pragma unroll
    for (int qi = 0; qi < L; ++qi) {
      float mx = sc[qi][0];
#pragma unroll
      for (int ki = 1; ki < L; ++ki) mx = fmaxf(mx, sc[qi][ki]);
      float sum = 0.f;
#pragma unroll
      for (int ki = 0; ki < L; ++ki) { sc[qi][ki] = expf(sc[qi][ki] - mx); sum += sc[qi][ki]; }
      float inv = 1.f / sum;
#pragma unroll
      for (int ki = 0; ki < L; ++ki) sc[qi][ki] *= inv;
    }
#pragma unroll
    for (int ki = 0; ki < L; ++ki) {
      float a = 0.f;
#pragma unroll
      for (int qi = 0; qi < L; ++qi) a += sc[qi][ki];
      atomicAdd(&keyavg[s * L + ki], a);            // sum over heads & queries
    }
#pragma unroll
    for (int qi = 0; qi < L; ++qi) {
#pragma clang loop unroll(disable)
      for (int e = 0; e < 16; ++e) {
        float c = 0.f;
#pragma unroll
        for (int ki = 0; ki < L; ++ki) c += sc[qi][ki] * base[ki * 192 + 128 + h * 16 + e];
        ctxH[(s * L + qi) * 64 + h * 16 + e] = (_Float16)c;
      }
    }
  }
  __syncthreads();
}

// ---------------- pre-norm encoder block with double-buffered TDM weight pipeline ----------------
// TDM schedule (wave0, in-order completion): issue Win->wA, Wout->wB; wait<=1 -> QKV(wA);
// issue W1->wA; wait<=1 -> WoutGEMM(wB); issue W2->wB; wait<=1 -> FFN1(wA); wait<=0 -> FFN2(wB)
template <int RPS, int TOKOFF, int L>
__device__ void encoder_block(float* resid, const BlockP& P, const _Float16* gw, bool tdm,
                              _Float16* a16, _Float16* wA, _Float16* wB,
                              float* qkv, float* keyavg) {
  constexpr int M = SPB * L;                            // 16 / 48 / 16
  int tid = threadIdx.x;
  if (tdm) {
    if (tid < 32) {
      tdm_load_w(gw,          lds_off(wA), 64, 192);    // Win  (192x64)
      tdm_load_w(gw + 12288,  lds_off(wB), 64, 64);     // Wout (64x64)
    }
  } else {
    stage_w(P.Win,  192 * 64, wA);
    stage_w(P.Wout, 64 * 64,  wB);
  }
  layernorm_rows<RPS, TOKOFF, L>(resid, M, P.ln1_g, P.ln1_b, a16);
  if (tdm && tid < 32) tdm_wait<1>();                   // Win landed
  __syncthreads();
  wgemm<0, 0, 0, 1>(a16, 64, M, wA, 64, 192, 64, P.bin,
                    qkv, 192, nullptr, 0, nullptr, nullptr);
  __syncthreads();                                      // all waves done reading wA
  if (tdm) { if (tid < 32) tdm_load_w(gw + 16384, lds_off(wA), 64, 256); }  // W1 (256x64)
  else stage_w(P.W1, 256 * 64, wA);
  attention<L>(qkv, keyavg, a16);                       // ctx f16 back into a16 (syncs inside)
  if (tdm && tid < 32) tdm_wait<1>();                   // Wout landed
  __syncthreads();
  wgemm<1, RPS, TOKOFF, L>(a16, 64, M, wB, 64, 64, 64, P.bout,
                           resid, 64, nullptr, 0, nullptr, nullptr);
  __syncthreads();                                      // all waves done reading wB
  if (tdm) { if (tid < 32) tdm_load_w(gw + 32768, lds_off(wB), 256, 64); } // W2 (64x256)
  else stage_w(P.W2, 64 * 256, wB);
  layernorm_rows<RPS, TOKOFF, L>(resid, M, P.ln2_g, P.ln2_b, a16);
  if (tdm && tid < 32) tdm_wait<1>();                   // W1 landed
  __syncthreads();
  _Float16* hbuf = a16 + 64 * 64;                       // FFN hidden (f16) region
  wgemm<2, 0, 0, 1>(a16, 64, M, wA, 64, 256, 64, P.b1,
                    nullptr, 0, hbuf, 256, nullptr, nullptr);
  if (tdm && tid < 32) tdm_wait<0>();                   // W2 landed
  __syncthreads();
  wgemm<1, RPS, TOKOFF, L>(hbuf, 256, M, wB, 256, 64, 256, P.b2,
                           resid, 64, nullptr, 0, nullptr, nullptr);
  __syncthreads();
}

// ---------------- fused AttentionPPO forward ----------------
__global__ __launch_bounds__(NTHREADS)
void attnppo_kernel(PpoArgs A, const _Float16* wf16, int useTdm) {
  __shared__ float    s_res[64 * 64];       // residual: 8 samples x 8 tokens x 64
  __shared__ _Float16 s_a16[64 * 256];      // f16 staging (A operands / ctx / FFN hidden)
  __shared__ _Float16 s_w16[2 * 16384];     // double-buffered f16 weights (wA | wB)
  __shared__ float    s_qkv[48 * 192];      // QKV scratch (also reused for head hiddens)
  __shared__ float    s_fus[16 * 64];       // fusion residual stream
  __shared__ float    s_pool[8 * 64];
  __shared__ float    s_kmac[16], s_kmic[48], s_kfus[16];

  int tid = threadIdx.x;
  int s0  = blockIdx.x * SPB;
  int Bn  = A.B;
  bool tdm = HAVE_TDM && useTdm;
  _Float16* wA = s_w16;
  _Float16* wB = s_w16 + 16384;

  // ---- stage tokW (TDM or fallback) and obs tokens as f16 A-matrix (64 x 64, zero-padded) ----
  if (tdm) { if (tid < 32) tdm_load_w(wf16 + WF_TOKW, lds_off(wA), 64, 64); }
  else stage_tokw(A.tokW, wA);
#pragma clang loop unroll_count(2)
  for (int i = tid; i < 64 * 64; i += NTHREADS) {
    int r = i >> 6, c = i & 63;
    int s = r >> 3, t = r & 7;
    int gs = s0 + s; if (gs >= Bn) gs = Bn - 1;
    float v = (c < 54) ? A.obs[(long)gs * 432 + t * 54 + c] : 0.f;
    s_a16[r * 64 + c] = (_Float16)v;
  }
  if (tdm && tid < 32) tdm_wait<0>();
  __syncthreads();
  // token embedding + positional (MODE 3)
  wgemm<3, 0, 0, 1>(s_a16, 64, 64, wA, 64, 64, 64, A.tokb,
                    s_res, 64, nullptr, 0, A.macro_pos, A.micro_pos);
  __syncthreads();

  // ---- macro (tokens 0-1) and micro (tokens 2-7) encoder blocks ----
  encoder_block<8, 0, 2>(s_res, A.mac, wf16 + WF_BLK(0), tdm, s_a16, wA, wB, s_qkv, s_kmac);
  encoder_block<8, 2, 6>(s_res, A.mic, wf16 + WF_BLK(1), tdm, s_a16, wA, wB, s_qkv, s_kmic);

  // ---- build fusion input: [macro_ctx, micro_ctx] + fusion_pos ----
#pragma clang loop unroll(disable)
  for (int i = tid; i < 16 * 64; i += NTHREADS) {
    int r = i >> 6, c = i & 63;
    int s = r >> 1, half = r & 1;
    float m;
    if (!half) m = 0.5f * (s_res[(s * 8 + 0) * 64 + c] + s_res[(s * 8 + 1) * 64 + c]);
    else {
      m = 0.f;
#pragma unroll
      for (int t = 2; t < 8; ++t) m += s_res[(s * 8 + t) * 64 + c];
      m *= (1.f / 6.f);
    }
    s_fus[r * 64 + c] = m + A.fusion_pos[half * 64 + c];
  }
  __syncthreads();

  encoder_block<2, 0, 2>(s_fus, A.fus, wf16 + WF_BLK(2), tdm, s_a16, wA, wB, s_qkv, s_kfus);

  // ---- pooled = mean over the 2 fusion rows ----
#pragma clang loop unroll(disable)
  for (int i = tid; i < 8 * 64; i += NTHREADS) {
    int s = i >> 6, c = i & 63;
    s_pool[i] = 0.5f * (s_fus[(2 * s) * 64 + c] + s_fus[(2 * s + 1) * 64 + c]);
  }
  __syncthreads();

  // ---- heads: relu(pooled@W1^T+b1) for actor & critic (VALU, tiny) ----
  float* ha = s_qkv;            // 8x64
  float* hc = s_qkv + 512;      // 8x64
#pragma clang loop unroll(disable)
  for (int i = tid; i < 1024; i += NTHREADS) {
    int net = i >> 9, rem = i & 511;
    int s = rem >> 6, j = rem & 63;
    const float* Wr = (net ? A.cW1 : A.aW1) + j * 64;
    float d = (net ? A.cb1 : A.ab1)[j];
    const float* x = s_pool + s * 64;
#pragma clang loop unroll(disable)
    for (int e = 0; e < 64; ++e) d += x[e] * Wr[e];
    (net ? hc : ha)[s * 64 + j] = fmaxf(d, 0.f);
  }
  __syncthreads();

  // ---- logits (B,4) and value (B,1) ----
  if (tid < SPB * 5) {
    int s = tid / 5, o = tid - s * 5;
    int gs = s0 + s;
    if (gs < Bn) {
      if (o < 4) {
        const float* Wr = A.aW2 + o * 64;
        const float* x  = ha + s * 64;
        float d = A.ab2[o];
#pragma clang loop unroll(disable)
        for (int e = 0; e < 64; ++e) d += x[e] * Wr[e];
        A.out[(long)gs * 4 + o] = d;
      } else {
        const float* Wr = A.cW2;
        const float* x  = hc + s * 64;
        float d = A.cb2[0];
#pragma clang loop unroll(disable)
        for (int e = 0; e < 64; ++e) d += x[e] * Wr[e];
        A.out[(long)Bn * 4 + gs] = d;
      }
    }
  }

  // ---- attn_compat (B,1,8,8): token importance broadcast over query dim ----
  if (tid >= 64 && tid < 128) {
    int idx = tid - 64;
    int s = idx >> 3, kk = idx & 7;
    int gs = s0 + s;
    if (gs < Bn) {
      float mimp = s_kfus[s * 2 + 0] * 0.125f;          // /(H*Lf)=/(4*2)
      float uimp = s_kfus[s * 2 + 1] * 0.125f;
      float msum = (s_kmac[s * 2 + 0] + s_kmac[s * 2 + 1]) * 0.125f;
      float usum = 0.f;
#pragma unroll
      for (int k = 0; k < 6; ++k) usum += s_kmic[s * 6 + k];
      usum *= (1.f / 24.f);
      float ti[8];
      float tsum = 0.f;
#pragma unroll
      for (int k = 0; k < 2; ++k) {
        float v = mimp * (s_kmac[s * 2 + k] * 0.125f) / (msum + 1e-8f);
        ti[k] = v; tsum += v;
      }
#pragma unroll
      for (int k = 0; k < 6; ++k) {
        float v = uimp * (s_kmic[s * 6 + k] * (1.f / 24.f)) / (usum + 1e-8f);
        ti[2 + k] = v; tsum += v;
      }
      float val = ti[kk] / (tsum + 1e-8f);
      float* dst = A.out + (long)Bn * 5 + (long)gs * 64;
#pragma clang loop unroll(disable)
      for (int q = 0; q < 8; ++q) dst[q * 8 + kk] = val;
    }
  }
}

// ---------------- pre-pass: convert all weights to f16 blob in d_ws ----------------
__global__ __launch_bounds__(NTHREADS)
void convert_w_kernel(PpoArgs A, _Float16* dst) {
  int tid = blockIdx.x * blockDim.x + threadIdx.x;
  int stride = gridDim.x * blockDim.x;
#pragma clang loop unroll(disable)
  for (int i = tid; i < 64 * 64; i += stride) {       // tokW zero-padded to 64x64
    int r = i >> 6, c = i & 63;
    dst[WF_TOKW + i] = (c < 54) ? (_Float16)A.tokW[r * 54 + c] : (_Float16)0.f;
  }
  const BlockP* Ps[3] = { &A.mac, &A.mic, &A.fus };
#pragma unroll
  for (int b = 0; b < 3; ++b) {
    const BlockP& P = *Ps[b];
    _Float16* base = dst + WF_BLK(b);
#pragma clang loop unroll(disable)
    for (int i = tid; i < 12288; i += stride) base[i]         = (_Float16)P.Win[i];
#pragma clang loop unroll(disable)
    for (int i = tid; i < 4096;  i += stride) base[12288 + i] = (_Float16)P.Wout[i];
#pragma clang loop unroll(disable)
    for (int i = tid; i < 16384; i += stride) base[16384 + i] = (_Float16)P.W1[i];
#pragma clang loop unroll(disable)
    for (int i = tid; i < 16384; i += stride) base[32768 + i] = (_Float16)P.W2[i];
  }
}

// ---------------- host launch ----------------
extern "C" void kernel_launch(void* const* d_in, const int* in_sizes, int n_in,
                              void* d_out, int out_size, void* d_ws, size_t ws_size,
                              hipStream_t stream) {
  (void)n_in; (void)out_size;
  PpoArgs a;
  a.obs        = (const float*)d_in[0];
  a.tokW       = (const float*)d_in[1];
  a.tokb       = (const float*)d_in[2];
  a.macro_pos  = (const float*)d_in[3];
  a.micro_pos  = (const float*)d_in[4];
  a.fusion_pos = (const float*)d_in[5];
  auto blk = [&](int base) {
    BlockP p;
    p.ln1_g = (const float*)d_in[base + 0];  p.ln1_b = (const float*)d_in[base + 1];
    p.Win   = (const float*)d_in[base + 2];  p.bin   = (const float*)d_in[base + 3];
    p.Wout  = (const float*)d_in[base + 4];  p.bout  = (const float*)d_in[base + 5];
    p.ln2_g = (const float*)d_in[base + 6];  p.ln2_b = (const float*)d_in[base + 7];
    p.W1    = (const float*)d_in[base + 8];  p.b1    = (const float*)d_in[base + 9];
    p.W2    = (const float*)d_in[base + 10]; p.b2    = (const float*)d_in[base + 11];
    return p;
  };
  a.mac = blk(6);
  a.mic = blk(18);
  a.fus = blk(30);
  a.aW1 = (const float*)d_in[42]; a.ab1 = (const float*)d_in[43];
  a.aW2 = (const float*)d_in[44]; a.ab2 = (const float*)d_in[45];
  a.cW1 = (const float*)d_in[46]; a.cb1 = (const float*)d_in[47];
  a.cW2 = (const float*)d_in[48]; a.cb2 = (const float*)d_in[49];
  a.out = (float*)d_out;
  a.B   = in_sizes[0] / 432;

  int useTdm = (ws_size >= (size_t)WF_TOTAL * sizeof(_Float16)) ? 1 : 0;
  _Float16* wf16 = (_Float16*)d_ws;
  if (useTdm)
    convert_w_kernel<<<dim3(64), dim3(NTHREADS), 0, stream>>>(a, wf16);

  int grid = (a.B + SPB - 1) / SPB;
  attnppo_kernel<<<dim3(grid), dim3(NTHREADS), 0, stream>>>(a, wf16, useTdm);
}
